// Trans_79602923864525
// MI455X (gfx1250) — matrix operand order repaired and checked by reading, not compiled, not run on hardware
//
#include <hip/hip_runtime.h>
#include <hip/hip_bf16.h>

// ---------------------------------------------------------------------------
// Problem constants (B, C, L) = (16, 128, 2048); feature dim D = 2C = 256.
// ---------------------------------------------------------------------------
#define BATCH 16
#define CDIM  128
#define LDIM  2048
#define DDIM  256

typedef __attribute__((ext_vector_type(16))) __bf16    v16bf;
typedef __attribute__((ext_vector_type(8)))  float     v8f;
typedef __attribute__((ext_vector_type(4)))  unsigned  u32x4;

union Frag {             // 16 bf16 = 8 dwords = two 16B vector loads
  v16bf v;
  unsigned u[8];
  u32x4 q[2];
};

__device__ __forceinline__ unsigned short f32_bits_bf16(float f) {
  unsigned u = __float_as_uint(f);
  u += 0x7FFFu + ((u >> 16) & 1u);          // round-to-nearest-even
  return (unsigned short)(u >> 16);
}

// ---------------------------------------------------------------------------
// Phase 1: q,k,v = W x + b  (fp32 math, tiny FLOP count) and bf16 packing:
//   Abuf[b][m][0:128]   = q[:,m]      Abuf[b][m][128:256] = pos_emb[m,:]
//   Bbuf[b][l][0:128]   = k[:,l]      Bbuf[b][l][128:256] = q[:,l]
//   Vbuf[b][c][l]       = v[c,l]
// ---------------------------------------------------------------------------
__global__ __launch_bounds__(256) void qkv_pack_kernel(
    const float* __restrict__ x,
    const float* __restrict__ Wq, const float* __restrict__ bq,
    const float* __restrict__ Wk, const float* __restrict__ bk,
    const float* __restrict__ Wv, const float* __restrict__ bv,
    const float* __restrict__ pos,
    unsigned short* __restrict__ Abuf,
    unsigned short* __restrict__ Bbuf,
    unsigned short* __restrict__ Vbuf)
{
  __shared__ float xs[CDIM * 64];
  const int b  = blockIdx.y;
  const int l0 = blockIdx.x * 64;
  const int t  = threadIdx.x;

  // stage x tile (coalesced along l)
  for (int i = t; i < CDIM * 64; i += 256) {
    const int ci = i >> 6, l = i & 63;
    xs[ci * 64 + l] = x[((size_t)b * CDIM + ci) * LDIM + l0 + l];
  }
  // pack positional embeddings into the query-side feature vector
  for (int i = t; i < 64 * CDIM; i += 256) {
    const int l = i >> 7, ci = i & 127;
    Abuf[((size_t)(b * LDIM + l0 + l)) * DDIM + CDIM + ci] =
        f32_bits_bf16(pos[(size_t)(l0 + l) * CDIM + ci]);
  }
  __syncthreads();

  const int l = t & 63, cog = t >> 6;      // 4 output-channel groups x 64 positions
  for (int co = cog; co < CDIM; co += 4) {
    float aq = bq[co], ak = bk[co], av = bv[co];
    const float* wq = Wq + (size_t)co * CDIM;
    const float* wk = Wk + (size_t)co * CDIM;
    const float* wv = Wv + (size_t)co * CDIM;
#pragma unroll 8
    for (int ci = 0; ci < CDIM; ++ci) {
      const float xv = xs[ci * 64 + l];
      aq = fmaf(wq[ci], xv, aq);
      ak = fmaf(wk[ci], xv, ak);
      av = fmaf(wv[ci], xv, av);
    }
    const size_t row = (size_t)(b * LDIM + l0 + l);
    const unsigned short qb = f32_bits_bf16(aq);
    Abuf[row * DDIM + co]        = qb;                  // query features
    Bbuf[row * DDIM + co]        = f32_bits_bf16(ak);   // key features: k
    Bbuf[row * DDIM + CDIM + co] = qb;                  // key features: q
    Vbuf[((size_t)(b * CDIM + co)) * LDIM + l0 + l] = f32_bits_bf16(av);
  }
}

// ---------------------------------------------------------------------------
// Phase 2: flash attention.  Block = 8 waves x 16 query rows = 128 rows.
// Each wave keeps its 16x256 A-fragments + 16x128 f32 accumulators resident,
// streams 64-key tiles through LDS, and uses v_wmma_f32_16x16x32_bf16 for
// both the score GEMM (K=256) and the output GEMM (K=64).
// ---------------------------------------------------------------------------
__global__ __launch_bounds__(256) void attn_kernel(
    const unsigned short* __restrict__ Abuf,
    const unsigned short* __restrict__ Bbuf,
    const unsigned short* __restrict__ Vbuf,
    float* __restrict__ out)
{
  __shared__ unsigned short Btile[64 * DDIM];      // 32 KB: key features [l][256]
  __shared__ unsigned short Vtile[CDIM * 64];      // 16 KB: values       [c][64]
  __shared__ unsigned short Ptile[8 * 16 * 64];    // 16 KB: per-wave P scratch

  const int b    = blockIdx.y;
  const int m0   = blockIdx.x * 128;
  const int t    = threadIdx.x;
  const int wave = t >> 5;
  const int lane = t & 31;
  const int col  = lane & 15;
  const int hv   = lane >> 4;                      // 0: rows 0-7 / K-low, 1: rows 8-15 / K-high

  // --- preload this wave's A fragments (16 rows x 256 features, bf16) -------
  // A-matrix VGPR layout (ISA 7.12.2): lane holds row M=lane&15;
  // half 0 -> K {0..7,16..23}, half 1 -> K {8..15,24..31} per 32-wide chunk.
  Frag afr[8];
  {
    const int mrowg = m0 + wave * 16 + col;
    const u32x4* ap = (const u32x4*)(Abuf + ((size_t)b * LDIM + mrowg) * DDIM);
    const int kbase = hv ? 8 : 0;
#pragma unroll
    for (int kk = 0; kk < 8; ++kk) {
      const int so = kk * 32 + kbase;              // offset in bf16 elements
      afr[kk].q[0] = ap[so >> 3];
      afr[kk].q[1] = ap[(so + 16) >> 3];
    }
  }

  float mrow[8], srow[8];
  v8f o[8];
#pragma unroll
  for (int r = 0; r < 8; ++r) { mrow[r] = -1e30f; srow[r] = 0.f; }
#pragma unroll
  for (int n = 0; n < 8; ++n) o[n] = (v8f){0.f,0.f,0.f,0.f,0.f,0.f,0.f,0.f};

  const unsigned short* Bg = Bbuf + (size_t)b * LDIM * DDIM;
  const unsigned short* Vg = Vbuf + (size_t)b * CDIM * LDIM;
  const int bk0 = hv ? 8 : 0;                      // B-matrix K base per half

  for (int lb = 0; lb < 32; ++lb) {
    const int lq = lb * 64;
    __syncthreads();
    // ---- cooperative loads: Btile is one contiguous 32 KB slab ------------
    {
      const u32x4* src = (const u32x4*)(Bg + (size_t)lq * DDIM);
      u32x4* dst = (u32x4*)Btile;
      for (int i = t; i < 2048; i += 256) dst[i] = src[i];
    }
    {
      u32x4* dst = (u32x4*)Vtile;
      for (int i = t; i < 1024; i += 256) {
        const int c = i >> 3, j = i & 7;
        dst[i] = ((const u32x4*)(Vg + (size_t)c * LDIM + lq))[j];
      }
    }
    if (lb < 31) {                                 // prefetch next key tile
      __builtin_prefetch(Bg + (size_t)(lq + 64) * DDIM, 0, 1);
      __builtin_prefetch(Vg + (size_t)(lq + 64), 0, 1);
    }
    __syncthreads();

    // ---- scores: S(16x64) = A(16x256) x B(256x64), 32 WMMAs ---------------
    v8f s[4];
#pragma unroll
    for (int tn = 0; tn < 4; ++tn) {
      v8f acc = (v8f){0.f,0.f,0.f,0.f,0.f,0.f,0.f,0.f};
      const u32x4* bp = (const u32x4*)(Btile + (tn * 16 + col) * DDIM);
#pragma unroll
      for (int kk = 0; kk < 8; ++kk) {
        Frag bf;
        const int so = kk * 32 + bk0;              // 8 bf16 at so, 8 at so+16
        bf.q[0] = bp[so >> 3];
        bf.q[1] = bp[(so >> 3) + 2];
        acc = __builtin_amdgcn_wmma_f32_16x16x32_bf16(
            false, afr[kk].v, false, bf.v, (short)0, acc, false, false);
      }
      s[tn] = acc;
    }

    // ---- online softmax (row stats per 16-lane half via shfl_xor) ---------
    unsigned short* Pw = Ptile + wave * 1024;
#pragma unroll
    for (int r = 0; r < 8; ++r) {
      float mx = fmaxf(fmaxf(s[0][r], s[1][r]), fmaxf(s[2][r], s[3][r]));
#pragma unroll
      for (int off = 8; off >= 1; off >>= 1) mx = fmaxf(mx, __shfl_xor(mx, off, 32));
      const float mnew = fmaxf(mrow[r], mx);
      const float corr = __expf(mrow[r] - mnew);
      mrow[r] = mnew;
      srow[r] *= corr;
#pragma unroll
      for (int n = 0; n < 8; ++n) o[n][r] *= corr;
      float rs = 0.f;
      const int mlocal = r + (hv ? 8 : 0);
#pragma unroll
      for (int tn = 0; tn < 4; ++tn) {
        const float p = __expf(s[tn][r] - mnew);
        rs += p;
        Pw[mlocal * 64 + tn * 16 + col] = f32_bits_bf16(p);   // re-layout via LDS
      }
#pragma unroll
      for (int off = 8; off >= 1; off >>= 1) rs += __shfl_xor(rs, off, 32);
      srow[r] += rs;
    }
    asm volatile("s_wait_dscnt 0" ::: "memory");   // wave-local LDS RAW fence

    // ---- O(16x128) += P(16x64) x V(64x128), 16 WMMAs ----------------------
#pragma unroll
    for (int kk2 = 0; kk2 < 2; ++kk2) {
      Frag pf;                                     // A-layout load of P
      const u32x4* pp =
          (const u32x4*)(Pw + col * 64 + kk2 * 32 + (hv ? 8 : 0));
      pf.q[0] = pp[0];
      pf.q[1] = pp[2];
#pragma unroll
      for (int n = 0; n < 8; ++n) {
        Frag vf;                                   // B-layout load of V
        const u32x4* vp =
            (const u32x4*)(Vtile + (n * 16 + col) * 64 + kk2 * 32 + bk0);
        vf.q[0] = vp[0];
        vf.q[1] = vp[2];
        o[n] = __builtin_amdgcn_wmma_f32_16x16x32_bf16(
            false, pf.v, false, vf.v, (short)0, o[n], false, false);
      }
    }
  }

  // ---- epilogue: out[b][c][m] = O[m][c] / rowsum ---------------------------
  float* og = out + (size_t)b * CDIM * LDIM;
  const int mbase = m0 + wave * 16;
#pragma unroll
  for (int n = 0; n < 8; ++n) {
    const int c = n * 16 + col;
#pragma unroll
    for (int r = 0; r < 8; ++r) {
      const int m = mbase + r + (hv ? 8 : 0);
      og[(size_t)c * LDIM + m] = o[n][r] / srow[r];
    }
  }
}

// ---------------------------------------------------------------------------
extern "C" void kernel_launch(void* const* d_in, const int* in_sizes, int n_in,
                              void* d_out, int out_size, void* d_ws, size_t ws_size,
                              hipStream_t stream) {
  const float* x   = (const float*)d_in[0];
  const float* Wq  = (const float*)d_in[1];
  const float* bq  = (const float*)d_in[2];
  const float* Wk  = (const float*)d_in[3];
  const float* bk  = (const float*)d_in[4];
  const float* Wv  = (const float*)d_in[5];
  const float* bv  = (const float*)d_in[6];
  const float* pos = (const float*)d_in[7];

  unsigned short* Abuf = (unsigned short*)d_ws;                       // 16 MB
  unsigned short* Bbuf = Abuf + (size_t)BATCH * LDIM * DDIM;          // 16 MB
  unsigned short* Vbuf = Bbuf + (size_t)BATCH * LDIM * DDIM;          //  8 MB

  dim3 g1(LDIM / 64, BATCH), blk(256);
  qkv_pack_kernel<<<g1, blk, 0, stream>>>(x, Wq, bq, Wk, bk, Wv, bv, pos,
                                          Abuf, Bbuf, Vbuf);

  dim3 g2(LDIM / 128, BATCH);
  attn_kernel<<<g2, blk, 0, stream>>>(Abuf, Bbuf, Vbuf, (float*)d_out);
}
